// SelfAttention_27599459844511
// MI455X (gfx1250) — matrix operand order repaired and checked by reading, not compiled, not run on hardware
//
#include <hip/hip_runtime.h>

typedef unsigned short u16;
typedef __attribute__((ext_vector_type(16))) __bf16 bf16x16;
typedef __attribute__((ext_vector_type(8)))  float  f32x8;

#define TDIM   2048
#define EDIM   1024
#define HEADS  16
#define HD     64
#define BATCH  2
#define MROWS  (BATCH * TDIM)   // 4096

// ---------------------------------------------------------------- helpers ---

__device__ __forceinline__ u16 f2bf(float x) {
  unsigned u = __float_as_uint(x);
  u += 0x7fffu + ((u >> 16) & 1u);          // round-to-nearest-even
  return (u16)(u >> 16);
}

union FragU { bf16x16 f; uint4 u[2]; };

// A-matrix 16x32 bf16 fragment (ISA 7.12.2): lane L holds row m=L%16,
// vgpr v holds k = 16*(v/4) + (L/16)*8 + 2*(v%4) + {0,1}
// => bytes [g*16, g*16+16) and [32+g*16, 32+g*16+16) of the lane's 32-elem row.
__device__ __forceinline__ bf16x16 frag_a_row(const u16* rowp, int g) {
  FragU r;
  r.u[0] = *(const uint4*)((const char*)rowp + g * 16);
  r.u[1] = *(const uint4*)((const char*)rowp + 32 + g * 16);
  return r.f;
}

// B-matrix 32x16 bf16 fragment: lane L holds col n=L%16, same k order as A:
// bytes [g*16, g*16+16) and [32+g*16, 32+g*16+16) of the n-major 32-elem row.
__device__ __forceinline__ bf16x16 frag_b_row(const u16* rowp, int g) {
  FragU r;
  r.u[0] = *(const uint4*)((const char*)rowp + g * 16);
  r.u[1] = *(const uint4*)((const char*)rowp + 32 + g * 16);
  return r.f;
}

__device__ __forceinline__ f32x8 wmma_bf16(bf16x16 a, bf16x16 b, f32x8 c) {
  // (neg_a, A, neg_b, B, c_mod, C, reuse_a, reuse_b)
  return __builtin_amdgcn_wmma_f32_16x16x32_bf16(false, a, false, b,
                                                 (short)0, c, false, false);
}

// CDNA5 async global->LDS copy (ASYNCcnt-tracked, no VGPR staging).
// VDST = 32-bit LDS byte address (low 32 bits of the generic pointer = AS3
// offset), VADDR = 64-bit global address, SADDR = off.
__device__ __forceinline__ void async_copy_b128(const u16* g, u16* l) {
  unsigned lds_addr = (unsigned)(uintptr_t)l;
  asm volatile("global_load_async_to_lds_b128 %0, %1, off"
               :: "v"(lds_addr), "v"(g)
               : "memory");
}
__device__ __forceinline__ void wait_async0() {
  asm volatile("s_wait_asynccnt 0x0" ::: "memory");
}

// ---------------------------------------------------------- prep kernels ---

__global__ void convert_bf16_kernel(const float* __restrict__ in,
                                    u16* __restrict__ out, int n) {
  int i = blockIdx.x * blockDim.x + threadIdx.x;
  if (i < n) out[i] = f2bf(in[i]);
}

// in: [rows][cols] fp32  ->  out: [cols][rows] bf16
__global__ void transpose_bf16_kernel(const float* __restrict__ in,
                                      u16* __restrict__ out, int rows, int cols) {
  int i = blockIdx.x * blockDim.x + threadIdx.x;
  if (i < rows * cols) {
    int r = i / cols, c = i % cols;
    out[(size_t)c * rows + r] = f2bf(in[i]);
  }
}

// ------------------------------------------------------------- GEMM core ---
// C[64 x 256] block tile, 256 threads = 8 waves (2 x 4), wave tile 32 x 64.
// A: [M][K] bf16 row-major.  BT: [N][K] bf16 (n-major).
// Double-buffered LDS tiles filled by global_load_async_to_lds_b128:
//   iter i: s_wait_asynccnt 0 (tile i landed) -> barrier ->
//           kick async loads for tile i+1 into buf^1 -> WMMA on buf (tile i).

__device__ __forceinline__ void gemm_core(const u16* __restrict__ A,
                                          const u16* __restrict__ BT,
                                          int K, int bm, int bn,
                                          u16* As, u16* Bs,   // As[2][64*32], Bs[2][256*32]
                                          f32x8 acc[2][4]) {
  const int tid  = threadIdx.x;
  const int lane = tid & 31;
  const int wave = tid >> 5;
  const int wm = wave & 1, wn = wave >> 1;
  const int g = lane >> 4, l16 = lane & 15;

  const int arow = tid >> 2;            // 0..63
  const int acol = (tid & 3) * 8;       // 0,8,16,24
  const u16* aG = A + (size_t)(bm * 64 + arow) * K + acol;
  const u16* bG = BT + (size_t)(bn * 256 + tid) * K;
  u16* aL0 = As + arow * 32 + acol;
  u16* bL0 = Bs + tid * 32;

  // prologue: tile 0 -> buffer 0
  async_copy_b128(aG, aL0);
#pragma unroll
  for (int j = 0; j < 4; ++j) async_copy_b128(bG + j * 8, bL0 + j * 8);

  const int nt = K / 32;
  for (int i = 0; i < nt; ++i) {
    wait_async0();          // my async writes for tile i are in LDS
    __syncthreads();        // everyone's are

    if (i + 1 < nt) {       // prefetch tile i+1 into the other buffer
      const int k1 = (i + 1) * 32;
      const int nb = (i + 1) & 1;
      async_copy_b128(aG + k1, aL0 + nb * (64 * 32));
#pragma unroll
      for (int j = 0; j < 4; ++j)
        async_copy_b128(bG + k1 + j * 8, bL0 + nb * (256 * 32) + j * 8);
    }

    const u16* as = As + (i & 1) * (64 * 32);
    const u16* bs = Bs + (i & 1) * (256 * 32);

    bf16x16 af[2], bf[4];
#pragma unroll
    for (int mf = 0; mf < 2; ++mf)
      af[mf] = frag_a_row(as + (wm * 32 + mf * 16 + l16) * 32, g);
#pragma unroll
    for (int nf = 0; nf < 4; ++nf)
      bf[nf] = frag_b_row(bs + (wn * 64 + nf * 16 + l16) * 32, g);
#pragma unroll
    for (int mf = 0; mf < 2; ++mf)
#pragma unroll
      for (int nf = 0; nf < 4; ++nf)
        acc[mf][nf] = wmma_bf16(af[mf], bf[nf], acc[mf][nf]);
  }
}

// ------------------------------------------------------ QKV projection -----
// C = Xbf @ W_attn + b_attn  -> scatter to Q/K/V [B,H,T,64] bf16.
// 1/sqrt(HD) = 0.125 folded into Q.

__global__ void __launch_bounds__(256)
gemm_qkv_kernel(const u16* __restrict__ X, const u16* __restrict__ WT,
                const float* __restrict__ bias,
                u16* __restrict__ Qb, u16* __restrict__ Kb, u16* __restrict__ Vb) {
  __shared__ __align__(16) u16 As[2][64 * 32];
  __shared__ __align__(16) u16 Bs[2][256 * 32];
  f32x8 acc[2][4];
#pragma unroll
  for (int i = 0; i < 2; ++i)
#pragma unroll
    for (int j = 0; j < 4; ++j) acc[i][j] = {};

  gemm_core(X, WT, EDIM, blockIdx.y, blockIdx.x, &As[0][0], &Bs[0][0], acc);

  const int lane = threadIdx.x & 31, wave = threadIdx.x >> 5;
  const int wm = wave & 1, wn = wave >> 1, g = lane >> 4, l16 = lane & 15;
#pragma unroll
  for (int mf = 0; mf < 2; ++mf)
#pragma unroll
    for (int nf = 0; nf < 4; ++nf) {
      const int ncol = blockIdx.x * 256 + wn * 64 + nf * 16 + l16;
      const float bv = bias[ncol];
      const int region = ncol >> 10;       // 0=Q 1=K 2=V
      const int e = ncol & 1023;
      const int h = e >> 6, d = e & 63;
      u16* dst = (region == 0) ? Qb : (region == 1 ? Kb : Vb);
      const float scale = (region == 0) ? 0.125f : 1.0f;
#pragma unroll
      for (int r = 0; r < 8; ++r) {
        const int row = blockIdx.y * 64 + wm * 32 + mf * 16 + g * 8 + r;
        const int bi = row >> 11, t = row & 2047;   // T = 2048
        const float v = (acc[mf][nf][r] + bv) * scale;
        dst[(((size_t)(bi * HEADS + h)) * TDIM + t) * HD + d] = f2bf(v);
      }
    }
}

// ---------------------------------------------------- flash attention ------
// grid: (T/64, B*H); block 128 = 4 waves; each wave owns 16 q rows and
// streams causal 32-key tiles with online softmax. Output: Y [B,T,E] bf16.
// K tile staged with async global->LDS copies (row-major == n-major B layout).

__global__ void __launch_bounds__(128)
attn_kernel(const u16* __restrict__ Q, const u16* __restrict__ K,
            const u16* __restrict__ V, u16* __restrict__ Y) {
  __shared__ __align__(16) u16 Ks[4][32 * 64];   // [key][d]  (n-major for QK^T)
  __shared__ __align__(16) u16 Vt[4][64 * 32];   // [d][key]  (n-major for PV)
  __shared__ __align__(16) u16 Ps[4][16 * 32];   // P staging (C-layout -> A-layout)

  const int lane = threadIdx.x & 31, wave = threadIdx.x >> 5;
  const int g = lane >> 4, l16 = lane & 15;
  const int bh = blockIdx.y;
  const size_t head = (size_t)bh * TDIM * HD;
  const int qbase = blockIdx.x * 64 + wave * 16;
  const int b = bh >> 4, h = bh & 15;

  // Q A-fragments (two 32-wide k windows over HD=64), resident in VGPRs.
  bf16x16 aq[2];
#pragma unroll
  for (int ks = 0; ks < 2; ++ks)
    aq[ks] = frag_a_row(Q + head + (size_t)(qbase + l16) * HD + ks * 32, g);

  float m_run[8], l_run[8];
  f32x8 o[4];
#pragma unroll
  for (int r = 0; r < 8; ++r) { m_run[r] = -__builtin_inff(); l_run[r] = 0.f; }
#pragma unroll
  for (int nf = 0; nf < 4; ++nf) o[nf] = {};

  u16* ksm = Ks[wave];
  u16* vsm = Vt[wave];
  u16* psm = Ps[wave];

  const int ntiles = qbase / 32 + 1;             // causal: only tiles <= q rows
  for (int kt = 0; kt < ntiles; ++kt) {
    const int kb = kt * 32;
    // --- K tile: async global->LDS, lane -> one key row (8 x 16B) ---
    {
      const u16* kp = K + head + (size_t)(kb + lane) * HD;
      u16* kd = ksm + lane * HD;
#pragma unroll
      for (int j = 0; j < 8; ++j) async_copy_b128(kp + j * 8, kd + j * 8);
    }
    // --- V tile: regular load + in-register transpose -> [d][key] ---
    {
      const uint4* vp = (const uint4*)(V + head + (size_t)(kb + lane) * HD);
      uint4 vv[8];
#pragma unroll
      for (int j = 0; j < 8; ++j) vv[j] = vp[j];
#pragma unroll
      for (int j = 0; j < 8; ++j)
#pragma unroll
        for (int q = 0; q < 4; ++q) {
          const unsigned w = (&vv[j].x)[q];
          const int d0 = j * 8 + q * 2;
          vsm[d0 * 32 + lane]       = (u16)(w & 0xffffu);
          vsm[(d0 + 1) * 32 + lane] = (u16)(w >> 16);
        }
    }
    wait_async0();                     // K tile landed in LDS
    __builtin_amdgcn_wave_barrier();   // keep fill before reads (in-order DS)

    // --- S = Q K^T (two 16-key column frags, HD contracted in two steps) ---
    f32x8 s[2];
#pragma unroll
    for (int nb = 0; nb < 2; ++nb) {
      f32x8 a = {};
#pragma unroll
      for (int ks = 0; ks < 2; ++ks) {
        bf16x16 kf = frag_b_row(ksm + (nb * 16 + l16) * HD + ks * 32, g);
        a = wmma_bf16(aq[ks], kf, a);
      }
      s[nb] = a;
    }

    // --- causal mask + online softmax (row stats per 16-lane half group) ---
#pragma unroll
    for (int r = 0; r < 8; ++r) {
      const int qrow = qbase + g * 8 + r;
      float v0 = (kb + l16 <= qrow)      ? s[0][r] : -__builtin_inff();
      float v1 = (kb + 16 + l16 <= qrow) ? s[1][r] : -__builtin_inff();
      float mx = fmaxf(v0, v1);
#pragma unroll
      for (int off = 1; off < 16; off <<= 1)
        mx = fmaxf(mx, __shfl_xor(mx, off, 16));
      const float mnew = fmaxf(m_run[r], mx);
      const float fac = __expf(m_run[r] - mnew);
      const float p0 = __expf(v0 - mnew);
      const float p1 = __expf(v1 - mnew);
      float rs = p0 + p1;
#pragma unroll
      for (int off = 1; off < 16; off <<= 1)
        rs += __shfl_xor(rs, off, 16);
      l_run[r] = l_run[r] * fac + rs;
      m_run[r] = mnew;
#pragma unroll
      for (int nf = 0; nf < 4; ++nf) o[nf][r] *= fac;
      psm[(g * 8 + r) * 32 + l16]      = f2bf(p0);
      psm[(g * 8 + r) * 32 + 16 + l16] = f2bf(p1);
    }
    __builtin_amdgcn_wave_barrier();

    // --- O += P V ---
    bf16x16 pf = frag_a_row(psm + l16 * 32, g);
#pragma unroll
    for (int nf = 0; nf < 4; ++nf) {
      bf16x16 vf = frag_b_row(vsm + (nf * 16 + l16) * 32, g);
      o[nf] = wmma_bf16(pf, vf, o[nf]);
    }
  }

  // --- normalize and store Y[b, t, h*64 + d] ---
#pragma unroll
  for (int r = 0; r < 8; ++r) {
    const float inv = 1.0f / l_run[r];
    const int t = qbase + g * 8 + r;
    const size_t rowoff = ((size_t)(b * TDIM + t)) * EDIM + h * HD;
#pragma unroll
    for (int nf = 0; nf < 4; ++nf)
      Y[rowoff + nf * 16 + l16] = f2bf(o[nf][r] * inv);
  }
}

// -------------------------------------------------------- output proj ------

__global__ void __launch_bounds__(256)
gemm_out_kernel(const u16* __restrict__ Yb, const u16* __restrict__ WT,
                const float* __restrict__ bias, float* __restrict__ out) {
  __shared__ __align__(16) u16 As[2][64 * 32];
  __shared__ __align__(16) u16 Bs[2][256 * 32];
  f32x8 acc[2][4];
#pragma unroll
  for (int i = 0; i < 2; ++i)
#pragma unroll
    for (int j = 0; j < 4; ++j) acc[i][j] = {};

  gemm_core(Yb, WT, EDIM, blockIdx.y, blockIdx.x, &As[0][0], &Bs[0][0], acc);

  const int lane = threadIdx.x & 31, wave = threadIdx.x >> 5;
  const int wm = wave & 1, wn = wave >> 1, g = lane >> 4, l16 = lane & 15;
#pragma unroll
  for (int mf = 0; mf < 2; ++mf)
#pragma unroll
    for (int nf = 0; nf < 4; ++nf) {
      const int ncol = blockIdx.x * 256 + wn * 64 + nf * 16 + l16;
      const float bv = bias[ncol];
#pragma unroll
      for (int r = 0; r < 8; ++r) {
        const int row = blockIdx.y * 64 + wm * 32 + mf * 16 + g * 8 + r;
        out[(size_t)row * EDIM + ncol] = acc[mf][nf][r] + bv;
      }
    }
}

// ------------------------------------------------------------- launcher ----

extern "C" void kernel_launch(void* const* d_in, const int* in_sizes, int n_in,
                              void* d_out, int out_size, void* d_ws, size_t ws_size,
                              hipStream_t stream) {
  (void)in_sizes; (void)n_in; (void)out_size; (void)ws_size;
  const float* x      = (const float*)d_in[0];
  const float* W_attn = (const float*)d_in[1];
  const float* b_attn = (const float*)d_in[2];
  const float* W_proj = (const float*)d_in[3];
  const float* b_proj = (const float*)d_in[4];
  float* out = (float*)d_out;

  char* ws = (char*)d_ws;                       // 48 MB layout
  u16* Xbf = (u16*)(ws);                        // [4096,1024]  8 MB
  u16* WaT = (u16*)(ws + (size_t)8  * 1048576); // [3072,1024]  6 MB
  u16* WpT = (u16*)(ws + (size_t)14 * 1048576); // [1024,1024]  2 MB
  u16* Qb  = (u16*)(ws + (size_t)16 * 1048576); // [B,H,T,64]   8 MB
  u16* Kb  = (u16*)(ws + (size_t)24 * 1048576); //              8 MB
  u16* Vb  = (u16*)(ws + (size_t)32 * 1048576); //              8 MB
  u16* Yb  = (u16*)(ws + (size_t)40 * 1048576); // [4096,1024]  8 MB

  const int nX = MROWS * EDIM;
  convert_bf16_kernel<<<nX / 256, 256, 0, stream>>>(x, Xbf, nX);
  const int nWa = EDIM * 3 * EDIM;
  transpose_bf16_kernel<<<(nWa + 255) / 256, 256, 0, stream>>>(W_attn, WaT, EDIM, 3 * EDIM);
  const int nWp = EDIM * EDIM;
  transpose_bf16_kernel<<<(nWp + 255) / 256, 256, 0, stream>>>(W_proj, WpT, EDIM, EDIM);

  gemm_qkv_kernel<<<dim3(12, 64), 256, 0, stream>>>(Xbf, WaT, b_attn, Qb, Kb, Vb);
  attn_kernel<<<dim3(TDIM / 64, BATCH * HEADS), 128, 0, stream>>>(Qb, Kb, Vb, Yb);
  gemm_out_kernel<<<dim3(4, 64), 256, 0, stream>>>(Yb, WpT, b_proj, out);
}
